// EGAT_33337536152097
// MI455X (gfx1250) — compile-verified, matrix operand order checked by reading, not run on hardware
//
#include <hip/hip_runtime.h>
#include <hip/hip_bf16.h>
#include <cmath>

// ---------------- problem constants ----------------
#define NN 50000          // nodes
#define NE 600000         // edges
#define NE2 300000        // edges used by edge classifier (E/2)
#define GG 128            // groups
#define SLOPE 0.2f

// ---------------- vector types (trivial, union-safe) ----------------
typedef __attribute__((ext_vector_type(16))) __bf16       v16bf;
typedef __attribute__((ext_vector_type(8)))  float        v8f;
typedef __attribute__((ext_vector_type(4)))  float        f4;
typedef __attribute__((ext_vector_type(4)))  unsigned int u4;
typedef __attribute__((__vector_size__(4 * sizeof(int)))) int v4i;

union Frag16 { v16bf v; unsigned short u[16]; };

// ---------------- CDNA5 async global->LDS copy (ASYNCcnt path) -------------
// clang-22 signature (from probe): (v4i* src_global, v4i* dst_lds, Ii, Ii)
#if __has_builtin(__builtin_amdgcn_global_load_async_to_lds_b128)
#define HAVE_ASYNC_LDS 1
#define GPTR(p) ((v4i*)(p))
#define LPTR(p) ((v4i*)(p))
#endif

static __device__ __forceinline__ void wait_async0() {
#if __has_builtin(__builtin_amdgcn_s_wait_asynccnt)
    __builtin_amdgcn_s_wait_asynccnt(0);
#else
    asm volatile("s_wait_asynccnt 0x0" ::: "memory");
#endif
}

// fp32 -> bf16 round-to-nearest-even
static __device__ __forceinline__ unsigned short f2bf(float x) {
    unsigned int u = __float_as_uint(x);
    u += 0x7FFFu + ((u >> 16) & 1u);
    return (unsigned short)(u >> 16);
}
static __device__ __forceinline__ float bf2f(unsigned short b) {
    return __uint_as_float(((unsigned int)b) << 16);
}
static __device__ __forceinline__ void pack8(unsigned short* dst, f4 a, f4 b) {
    dst[0] = f2bf(a.x); dst[1] = f2bf(a.y); dst[2] = f2bf(a.z); dst[3] = f2bf(a.w);
    dst[4] = f2bf(b.x); dst[5] = f2bf(b.y); dst[6] = f2bf(b.z); dst[7] = f2bf(b.w);
}

// ---- WMMA fragment loaders ----
// A fragment (16x32 bf16): lane L -> row (L&15); K base (L>=16 ? 8 : 0);
// elems 0..7 -> K=kb..kb+7, elems 8..15 -> K=kb+16..kb+23  (two 16B loads).
static __device__ __forceinline__ v16bf frag_A_lds(const unsigned short* tile, int lda,
                                                   int row0, int k0) {
    int lane = threadIdx.x & 31;
    int r  = row0 + (lane & 15);
    int kb = k0 + ((lane >> 4) << 3);
    const unsigned short* p = tile + r * lda + kb;
    Frag16 f;
    *(u4*)&f.u[0] = *(const u4*)(p);
    *(u4*)&f.u[8] = *(const u4*)(p + 16);
    return f.v;
}
// B fragment (32x16 bf16) read from a row-major W[N][K] bf16 weight: lane's
// column c == weight row c, so the K stripe is a contiguous pair of 16B loads.
static __device__ __forceinline__ v16bf frag_B_w(const unsigned short* W, int ldk,
                                                 int col0, int k0) {
    int lane = threadIdx.x & 31;
    int c  = col0 + (lane & 15);
    int kb = k0 + ((lane >> 4) << 3);
    const unsigned short* p = W + (size_t)c * ldk + kb;
    Frag16 f;
    *(u4*)&f.u[0] = *(const u4*)(p);
    *(u4*)&f.u[8] = *(const u4*)(p + 16);
    return f.v;
}

#define WMMA_BF16(A, B, C) \
    __builtin_amdgcn_wmma_f32_16x16x32_bf16(false, (A), false, (B), (short)0, (C), false, false)

// ---------------- small utility kernels ----------------
__global__ void fill_f32(float* p, float v, int n) {
    int i = blockIdx.x * 256 + threadIdx.x;
    if (i < n) p[i] = v;
}
__global__ void cvt_bf16(const float* __restrict__ src, unsigned short* __restrict__ dst, int n) {
    int i = blockIdx.x * 256 + threadIdx.x;
    if (i < n) dst[i] = f2bf(src[i]);
}
static __device__ __forceinline__ void atomicMaxFloat(float* addr, float val) {
    if (val >= 0.f) atomicMax((int*)addr, __float_as_int(val));
    else            atomicMin((unsigned int*)addr, __float_as_uint(val));
}

// ---------------- generic fused GEMM:  Y[M x 128] = act(X[M x K] Wt + b) ------
// X can be a concatenation of A1 (first K1 cols) and A2 (remaining cols).
// Writes fp32 C and/or a bf16 shadow Cbf (for later gathers). N fixed = 128.
__global__ __launch_bounds__(256) void gemm_bf16(
    const float* __restrict__ A1, const float* __restrict__ A2, int K1, int K,
    const unsigned short* __restrict__ Wb, const float* __restrict__ bias,
    float* __restrict__ C, unsigned short* __restrict__ Cbf, int M, int act)
{
    __shared__ __align__(16) unsigned short As[64][40];   // 64 x 32 bf16, padded
    const int t = threadIdx.x, wid = t >> 5, lane = t & 31;
    const int rows16 = (wid & 3) << 4;          // wave's 16-row block
    const int col0   = (wid >> 2) << 6;         // wave's 64-col strip
    const int m0 = blockIdx.x * 64;
    const int row = t >> 2, cseg = t & 3;
    const int K2 = K - K1;

    v8f acc[4];
    for (int c = 0; c < 4; ++c)
        for (int i = 0; i < 8; ++i) acc[c][i] = 0.f;

    for (int k0 = 0; k0 < K; k0 += 32) {
        __syncthreads();
        {   // stage 64x32 fp32 -> bf16 into LDS
            int gm = m0 + row;
            int kc = k0 + (cseg << 3);
            f4 f0 = {0.f, 0.f, 0.f, 0.f}, f1 = {0.f, 0.f, 0.f, 0.f};
            if (gm < M) {
                const float* src = (kc < K1) ? (A1 + (size_t)gm * K1 + kc)
                                             : (A2 + (size_t)gm * K2 + (kc - K1));
                f0 = *(const f4*)(src);
                f1 = *(const f4*)(src + 4);
            }
            __align__(16) unsigned short tmp[8];
            pack8(tmp, f0, f1);
            *(u4*)&As[row][cseg << 3] = *(u4*)tmp;
        }
        __syncthreads();
        // batch all B loads so the load clause overlaps the WMMA chain
        v16bf a  = frag_A_lds(&As[0][0], 40, rows16, 0);
        v16bf b0 = frag_B_w(Wb, K, col0 +  0, k0);
        v16bf b1 = frag_B_w(Wb, K, col0 + 16, k0);
        v16bf b2 = frag_B_w(Wb, K, col0 + 32, k0);
        v16bf b3 = frag_B_w(Wb, K, col0 + 48, k0);
        acc[0] = WMMA_BF16(a, b0, acc[0]);
        acc[1] = WMMA_BF16(a, b1, acc[1]);
        acc[2] = WMMA_BF16(a, b2, acc[2]);
        acc[3] = WMMA_BF16(a, b3, acc[3]);
    }
#pragma unroll
    for (int c = 0; c < 4; ++c) {
        int n = col0 + (c << 4) + (lane & 15);
        float bv = bias ? bias[n] : 0.f;
#pragma unroll
        for (int v = 0; v < 8; ++v) {
            int gm = m0 + rows16 + v + ((lane >> 4) << 3);
            if (gm < M) {
                float val = acc[c][v] + bv;
                if (act) val = val > 0.f ? val : 0.f;
                if (C)   C[(size_t)gm * 128 + n] = val;
                if (Cbf) Cbf[(size_t)gm * 128 + n] = f2bf(val);
            }
        }
    }
}

// ---------------- attention logit pass (8 lanes / edge) ----------------
__global__ __launch_bounds__(256) void edge_logit(
    const unsigned short* __restrict__ xbf,
    const float* __restrict__ eaf, const unsigned short* __restrict__ eab,
    const float* __restrict__ attn,           // (256 + Fe)
    const int* __restrict__ srcI, const int* __restrict__ dstI,
    float* __restrict__ dmax, float* __restrict__ logits)
{
    int gid = blockIdx.x * 256 + threadIdx.x;
    int e = gid >> 3, g = gid & 7;
    if (e >= NE) return;
    int s = srcI[e], d = dstI[e];
    float acc = 0.f;
    const unsigned short* xi = xbf + (size_t)d * 128 + (g << 4);
    const unsigned short* xj = xbf + (size_t)s * 128 + (g << 4);
    const float* ai = attn + (g << 4);
    const float* aj = attn + 128 + (g << 4);
    __align__(16) unsigned short ti[16], tj[16];
    *(u4*)&ti[0] = *(const u4*)(xi);  *(u4*)&ti[8] = *(const u4*)(xi + 8);
    *(u4*)&tj[0] = *(const u4*)(xj);  *(u4*)&tj[8] = *(const u4*)(xj + 8);
#pragma unroll
    for (int k = 0; k < 16; ++k) acc += ai[k] * bf2f(ti[k]) + aj[k] * bf2f(tj[k]);
    if (eaf) {                                   // layer 0: 64 fp32 feats
        const float* er = eaf + (size_t)e * 64 + (g << 3);
        const float* ae = attn + 256 + (g << 3);
#pragma unroll
        for (int k = 0; k < 8; ++k) acc += ae[k] * er[k];
    } else {                                     // layers 1..: 128 bf16 feats
        const unsigned short* er = eab + (size_t)e * 128 + (g << 4);
        const float* ae = attn + 256 + (g << 4);
        __align__(16) unsigned short te[16];
        *(u4*)&te[0] = *(const u4*)(er);  *(u4*)&te[8] = *(const u4*)(er + 8);
#pragma unroll
        for (int k = 0; k < 16; ++k) acc += ae[k] * bf2f(te[k]);
    }
    acc += __shfl_xor(acc, 1);
    acc += __shfl_xor(acc, 2);
    acc += __shfl_xor(acc, 4);
    if (g == 0) {
        float l = acc > 0.f ? acc : SLOPE * acc;   // leaky relu
        logits[e] = l;
        atomicMaxFloat(&dmax[d], l);
    }
}

__global__ void edge_exp(const float* __restrict__ dmax, float* __restrict__ logits,
                         float* __restrict__ dsum, const int* __restrict__ dstI)
{
    int e = blockIdx.x * 256 + threadIdx.x;
    if (e >= NE) return;
    int d = dstI[e];
    float m = dmax[d];
    if (m < -1e30f) m = 0.f;
    float ex = expf(logits[e] - m);
    logits[e] = ex;
    atomicAdd(&dsum[d], ex);
}

// ---------------- message GEMM + scatter-add aggregation ----------------
__global__ __launch_bounds__(256) void edge_msg(
    const unsigned short* __restrict__ xbf,
    const float* __restrict__ eaf, const unsigned short* __restrict__ eab, int Fe,
    const unsigned short* __restrict__ W3b,
    const float* __restrict__ alpha_ex, const float* __restrict__ dsum,
    const int* __restrict__ srcI, const int* __restrict__ dstI,
    float* __restrict__ aggr)
{
    __shared__ __align__(16) unsigned short Es[64][264];  // 64 edges x K<=256 bf16
    __shared__ float scaleS[64];
    __shared__ int   dstS[64];
    const int t = threadIdx.x, wid = t >> 5, lane = t & 31;
    const int e0 = blockIdx.x * 64;
    const int K = 128 + Fe;

    if (t < 64) {
        int e = e0 + t;
        if (e < NE) {
            int d = dstI[e];
            dstS[t] = d;
            scaleS[t] = alpha_ex[e] / (dsum[d] + 1e-16f);
        } else { dstS[t] = 0; scaleS[t] = 0.f; }
    }
    {   // stage [ x_j | edge_attr ] rows (bf16)
        int row = t >> 2, cseg = t & 3;
        int e = e0 + row;
        bool valid = e < NE;
#ifdef HAVE_ASYNC_LDS
        // Async DMA path: bf16 rows go global->LDS without touching VGPRs.
        // Out-of-range rows are clamped to a valid address; their scale is 0.
        int ec = valid ? e : NE - 1;
        int s = srcI[ec];
        const unsigned short* xr = xbf + (size_t)s * 128;
        for (int ch = cseg; ch < 16; ch += 4)
            __builtin_amdgcn_global_load_async_to_lds_b128(
                GPTR(xr + (ch << 3)), LPTR(&Es[row][ch << 3]), 0, 0);
        if (eaf) {   // layer 0: fp32 feats need VALU conversion -> sync path
            const float* er = eaf + (size_t)ec * 64;
            for (int ch = cseg; ch < 8; ch += 4) {
                f4 f0 = *(const f4*)(er + (ch << 3));
                f4 f1 = *(const f4*)(er + (ch << 3) + 4);
                __align__(16) unsigned short tmp[8];
                pack8(tmp, f0, f1);
                *(u4*)&Es[row][128 + (ch << 3)] = *(u4*)tmp;
            }
        } else {
            const unsigned short* er = eab + (size_t)ec * 128;
            for (int ch = cseg; ch < 16; ch += 4)
                __builtin_amdgcn_global_load_async_to_lds_b128(
                    GPTR(er + (ch << 3)), LPTR(&Es[row][128 + (ch << 3)]), 0, 0);
        }
        wait_async0();
#else
        int s = valid ? srcI[e] : 0;
        const unsigned short* xr = xbf + (size_t)s * 128;
        u4 z = {0u, 0u, 0u, 0u};
        for (int ch = cseg; ch < 16; ch += 4)
            *(u4*)&Es[row][ch << 3] = valid ? *(const u4*)(xr + (ch << 3)) : z;
        if (eaf) {
            const float* er = eaf + (size_t)e * 64;
            for (int ch = cseg; ch < 8; ch += 4) {
                f4 f0 = {0.f,0.f,0.f,0.f}, f1 = {0.f,0.f,0.f,0.f};
                if (valid) { f0 = *(const f4*)(er + (ch << 3)); f1 = *(const f4*)(er + (ch << 3) + 4); }
                __align__(16) unsigned short tmp[8];
                pack8(tmp, f0, f1);
                *(u4*)&Es[row][128 + (ch << 3)] = *(u4*)tmp;
            }
        } else {
            const unsigned short* er = eab + (size_t)e * 128;
            for (int ch = cseg; ch < 16; ch += 4)
                *(u4*)&Es[row][128 + (ch << 3)] = valid ? *(const u4*)(er + (ch << 3)) : z;
        }
#endif
    }
    __syncthreads();

    const int rows16 = (wid & 3) << 4;
    const int col0   = (wid >> 2) << 6;
    v8f acc[4];
    for (int c = 0; c < 4; ++c)
        for (int i = 0; i < 8; ++i) acc[c][i] = 0.f;
    for (int k0 = 0; k0 < K; k0 += 32) {
        v16bf a  = frag_A_lds(&Es[0][0], 264, rows16, k0);
        v16bf b0 = frag_B_w(W3b, K, col0 +  0, k0);
        v16bf b1 = frag_B_w(W3b, K, col0 + 16, k0);
        v16bf b2 = frag_B_w(W3b, K, col0 + 32, k0);
        v16bf b3 = frag_B_w(W3b, K, col0 + 48, k0);
        acc[0] = WMMA_BF16(a, b0, acc[0]);
        acc[1] = WMMA_BF16(a, b1, acc[1]);
        acc[2] = WMMA_BF16(a, b2, acc[2]);
        acc[3] = WMMA_BF16(a, b3, acc[3]);
    }
#pragma unroll
    for (int c = 0; c < 4; ++c) {
        int n = col0 + (c << 4) + (lane & 15);
#pragma unroll
        for (int v = 0; v < 8; ++v) {
            int ml = rows16 + v + ((lane >> 4) << 3);
            float sc = scaleS[ml];
            if (sc != 0.f)
                atomicAdd(&aggr[(size_t)dstS[ml] * 128 + n], acc[c][v] * sc);
        }
    }
}

// ---------------- node classifier final dot (8 lanes / node) ----------------
__global__ __launch_bounds__(256) void node_logit(
    const float* __restrict__ nh, const float* __restrict__ w1,
    const float* __restrict__ b1, float* __restrict__ outp)
{
    int gid = blockIdx.x * 256 + threadIdx.x;
    int i = gid >> 3, g = gid & 7;
    if (i >= NN) return;
    const float* r = nh + (size_t)i * 128 + (g << 4);
    const float* w = w1 + (g << 4);
    float acc = 0.f;
#pragma unroll
    for (int k = 0; k < 16; k += 4) {
        f4 f = *(const f4*)(r + k);
        f4 ww = *(const f4*)(w + k);
        acc += f.x * ww.x + f.y * ww.y + f.z * ww.z + f.w * ww.w;
    }
    acc += __shfl_xor(acc, 1);
    acc += __shfl_xor(acc, 2);
    acc += __shfl_xor(acc, 4);
    if (g == 0) outp[i] = acc + b1[0];
}

// ---------------- edge classifier: [|xa-xb|, xa+xb] @ ec0^T -> relu -> ec1 ----
__global__ __launch_bounds__(256) void edge_cls(
    const float* __restrict__ X, const int* __restrict__ ei,
    const unsigned short* __restrict__ W0b, const float* __restrict__ b0,
    const float* __restrict__ w1, const float* __restrict__ b1,
    float* __restrict__ outp)
{
    __shared__ __align__(16) unsigned short Xe[64][264];
    __shared__ float logitAcc[64];
    const int t = threadIdx.x, wid = t >> 5, lane = t & 31;
    const int j0 = blockIdx.x * 64;
    if (t < 64) logitAcc[t] = 0.f;
    {   // build x_edge rows in bf16
        int row = t >> 2, cseg = t & 3;
        int j = j0 + row;
        bool valid = j < NE2;
        int a = valid ? ei[2 * j] : 0;
        int b = valid ? ei[NE + 2 * j] : 0;
        const float* xa = X + (size_t)a * 128;
        const float* xb = X + (size_t)b * 128;
#pragma unroll
        for (int cc = 0; cc < 4; ++cc) {
            int col = (cseg << 5) + (cc << 3);
            f4 fa0 = {0.f,0.f,0.f,0.f}, fa1 = fa0, fb0 = fa0, fb1 = fa0;
            if (valid) {
                fa0 = *(const f4*)(xa + col); fa1 = *(const f4*)(xa + col + 4);
                fb0 = *(const f4*)(xb + col); fb1 = *(const f4*)(xb + col + 4);
            }
            f4 d0 = {fabsf(fa0.x-fb0.x), fabsf(fa0.y-fb0.y), fabsf(fa0.z-fb0.z), fabsf(fa0.w-fb0.w)};
            f4 d1 = {fabsf(fa1.x-fb1.x), fabsf(fa1.y-fb1.y), fabsf(fa1.z-fb1.z), fabsf(fa1.w-fb1.w)};
            f4 s0 = {fa0.x+fb0.x, fa0.y+fb0.y, fa0.z+fb0.z, fa0.w+fb0.w};
            f4 s1 = {fa1.x+fb1.x, fa1.y+fb1.y, fa1.z+fb1.z, fa1.w+fb1.w};
            __align__(16) unsigned short td[8], ts[8];
            pack8(td, d0, d1);
            pack8(ts, s0, s1);
            *(u4*)&Xe[row][col]       = *(u4*)td;
            *(u4*)&Xe[row][128 + col] = *(u4*)ts;
        }
    }
    __syncthreads();
    const int rows16 = (wid & 3) << 4;
    const int colb   = (wid >> 2) << 7;     // 0 or 128
    v8f acc[8];
    for (int c = 0; c < 8; ++c)
        for (int i = 0; i < 8; ++i) acc[c][i] = 0.f;
    for (int k0 = 0; k0 < 256; k0 += 32) {
        v16bf a = frag_A_lds(&Xe[0][0], 264, rows16, k0);
#pragma unroll
        for (int c = 0; c < 8; c += 2) {
            v16bf b0 = frag_B_w(W0b, 256, colb + (c << 4), k0);
            v16bf b1 = frag_B_w(W0b, 256, colb + ((c + 1) << 4), k0);
            acc[c]     = WMMA_BF16(a, b0, acc[c]);
            acc[c + 1] = WMMA_BF16(a, b1, acc[c + 1]);
        }
    }
#pragma unroll
    for (int c = 0; c < 8; ++c) {
        int n = colb + (c << 4) + (lane & 15);
        float bb = b0[n], ww = w1[n];
#pragma unroll
        for (int v = 0; v < 8; ++v) {
            int ml = rows16 + v + ((lane >> 4) << 3);
            float val = acc[c][v] + bb;
            val = val > 0.f ? val : 0.f;
            atomicAdd(&logitAcc[ml], val * ww);   // ds_add_f32 per-row reduce
        }
    }
    __syncthreads();
    if (t < 64) {
        int j = j0 + t;
        if (j < NE2) outp[j] = logitAcc[t] + b1[0];
    }
}

// ---------------- group softmax (3 atomic passes) ----------------
static __device__ __forceinline__ int entry_group(int t, const int* batch, const int* ei) {
    return (t < NN) ? batch[t] : batch[ei[2 * (t - NN)]];
}
__global__ void grp_max(const float* __restrict__ outp, const int* __restrict__ batch,
                        const int* __restrict__ ei, float* __restrict__ gmax) {
    int t = blockIdx.x * 256 + threadIdx.x;
    if (t >= NN + NE2) return;
    atomicMaxFloat(&gmax[entry_group(t, batch, ei)], outp[t]);
}
__global__ void grp_exp(float* __restrict__ outp, const int* __restrict__ batch,
                        const int* __restrict__ ei, const float* __restrict__ gmax,
                        float* __restrict__ gsum) {
    int t = blockIdx.x * 256 + threadIdx.x;
    if (t >= NN + NE2) return;
    int g = entry_group(t, batch, ei);
    float m = gmax[g];
    if (m < -1e30f) m = 0.f;
    float ex = expf(outp[t] - m);
    outp[t] = ex;
    atomicAdd(&gsum[g], ex);
}
__global__ void grp_norm(float* __restrict__ outp, const int* __restrict__ batch,
                         const int* __restrict__ ei, const float* __restrict__ gsum) {
    int t = blockIdx.x * 256 + threadIdx.x;
    if (t >= NN + NE2) return;
    outp[t] /= (gsum[entry_group(t, batch, ei)] + 1e-16f);
}

// ---------------- host orchestration ----------------
static inline int cdiv(int a, int b) { return (a + b - 1) / b; }

extern "C" void kernel_launch(void* const* d_in, const int* in_sizes, int n_in,
                              void* d_out, int out_size, void* d_ws, size_t ws_size,
                              hipStream_t stream)
{
    (void)in_sizes; (void)n_in; (void)out_size; (void)ws_size;
    // jax pytree leaf order: top-level insertion order, params dict keys sorted.
    const float* x      = (const float*)d_in[0];
    const int*   eidx   = (const int*)d_in[1];       // [2][E]
    const float* eattr  = (const float*)d_in[2];     // [E][64]
    const int*   ibatch = (const int*)d_in[3];
    const float* ec0_b  = (const float*)d_in[4];
    const float* ec0_w  = (const float*)d_in[5];     // [256][256]
    const float* ec1_b  = (const float*)d_in[6];
    const float* ec1_w  = (const float*)d_in[7];     // [1][256]
    const float* Wm1[3]   = {(const float*)d_in[8],  (const float*)d_in[15], (const float*)d_in[22]};
    const float* Wm3[3]   = {(const float*)d_in[9],  (const float*)d_in[16], (const float*)d_in[23]};
    const float* attnW[3] = {(const float*)d_in[10], (const float*)d_in[17], (const float*)d_in[24]};
    const float* m0b[3]   = {(const float*)d_in[11], (const float*)d_in[18], (const float*)d_in[25]};
    const float* m0w[3]   = {(const float*)d_in[12], (const float*)d_in[19], (const float*)d_in[26]};
    const float* m1b[3]   = {(const float*)d_in[13], (const float*)d_in[20], (const float*)d_in[27]};
    const float* m1w[3]   = {(const float*)d_in[14], (const float*)d_in[21], (const float*)d_in[28]};
    const float* lin_b = (const float*)d_in[29];
    const float* lin_w = (const float*)d_in[30];     // [128][64]
    const float* nc0_b = (const float*)d_in[31];
    const float* nc0_w = (const float*)d_in[32];
    const float* nc1_b = (const float*)d_in[33];
    const float* nc1_w = (const float*)d_in[34];

    const int* srcI = eidx;
    const int* dstI = eidx + NE;

    // workspace carve-up
    char* ws = (char*)d_ws;
    size_t off = 0;
    auto carve = [&](size_t bytes) -> char* {
        char* r = ws + off;
        off += (bytes + 255) & ~(size_t)255;
        return r;
    };
    float* bufA  = (float*)carve((size_t)NN * 128 * 4);
    float* bufB  = (float*)carve((size_t)NN * 128 * 4);
    float* xh    = (float*)carve((size_t)NN * 128 * 4);
    float* aggr  = (float*)carve((size_t)NN * 128 * 4);
    float* hmid  = (float*)carve((size_t)NN * 128 * 4);
    unsigned short* xbf   = (unsigned short*)carve((size_t)NN * 128 * 2);
    unsigned short* neaBf = (unsigned short*)carve((size_t)NE * 128 * 2);
    float* alphaB = (float*)carve((size_t)NE * 4);
    float* dmax   = (float*)carve((size_t)NN * 4);
    float* dsum   = (float*)carve((size_t)NN * 4);
    float* gmax   = (float*)carve((size_t)GG * 4);
    float* gsum   = (float*)carve((size_t)GG * 4);
    unsigned short* w_lin = (unsigned short*)carve((size_t)128 * 64 * 2);
    unsigned short* w_m1[3], *w_m3[3], *w_p0[3], *w_p1[3];
    int k3[3] = {192, 256, 256};
    for (int l = 0; l < 3; ++l) {
        w_m1[l] = (unsigned short*)carve((size_t)128 * 128 * 2);
        w_m3[l] = (unsigned short*)carve((size_t)128 * k3[l] * 2);
        w_p0[l] = (unsigned short*)carve((size_t)128 * 256 * 2);
        w_p1[l] = (unsigned short*)carve((size_t)128 * 128 * 2);
    }
    unsigned short* w_nc0 = (unsigned short*)carve((size_t)128 * 128 * 2);
    unsigned short* w_ec0 = (unsigned short*)carve((size_t)256 * 256 * 2);

    // ---- convert weights to bf16 once per call ----
    auto cvt = [&](const float* s, unsigned short* d, int n) {
        cvt_bf16<<<cdiv(n, 256), 256, 0, stream>>>(s, d, n);
    };
    cvt(lin_w, w_lin, 128 * 64);
    for (int l = 0; l < 3; ++l) {
        cvt(Wm1[l], w_m1[l], 128 * 128);
        cvt(Wm3[l], w_m3[l], 128 * k3[l]);
        cvt(m0w[l], w_p0[l], 128 * 256);
        cvt(m1w[l], w_p1[l], 128 * 128);
    }
    cvt(nc0_w, w_nc0, 128 * 128);
    cvt(ec0_w, w_ec0, 256 * 256);

    // ---- new_edge_attr (bf16 only) = edge_attr @ lin_w^T + lin_b ----
    gemm_bf16<<<cdiv(NE, 64), 256, 0, stream>>>(eattr, nullptr, 64, 64, w_lin, lin_b,
                                                nullptr, neaBf, NE, 0);

    // ---- EGAT layers ----
    const float* xcur = x;
    float* outbuf[2] = {bufA, bufB};
    for (int l = 0; l < 3; ++l) {
        int Fe = (l == 0) ? 64 : 128;
        const float*          eaf = (l == 0) ? eattr : nullptr;
        const unsigned short* eab = (l == 0) ? nullptr : neaBf;

        // x' = x @ Wm1^T  (fp32 + bf16 shadow)
        gemm_bf16<<<cdiv(NN, 64), 256, 0, stream>>>(xcur, nullptr, 128, 128, w_m1[l],
                                                    nullptr, xh, xbf, NN, 0);
        // segment softmax over dst
        fill_f32<<<cdiv(NN, 256), 256, 0, stream>>>(dmax, -INFINITY, NN);
        fill_f32<<<cdiv(NN, 256), 256, 0, stream>>>(dsum, 0.f, NN);
        edge_logit<<<cdiv(NE * 8, 256), 256, 0, stream>>>(xbf, eaf, eab, attnW[l],
                                                          srcI, dstI, dmax, alphaB);
        edge_exp<<<cdiv(NE, 256), 256, 0, stream>>>(dmax, alphaB, dsum, dstI);
        // weighted message aggregation
        fill_f32<<<cdiv(NN * 128, 256), 256, 0, stream>>>(aggr, 0.f, NN * 128);
        edge_msg<<<cdiv(NE, 64), 256, 0, stream>>>(xbf, eaf, eab, Fe, w_m3[l],
                                                   alphaB, dsum, srcI, dstI, aggr);
        // h = relu([aggr, x'] @ mlp0^T + b0);  out = h @ mlp1^T + b1
        gemm_bf16<<<cdiv(NN, 64), 256, 0, stream>>>(aggr, xh, 128, 256, w_p0[l],
                                                    m0b[l], hmid, nullptr, NN, 1);
        gemm_bf16<<<cdiv(NN, 64), 256, 0, stream>>>(hmid, nullptr, 128, 128, w_p1[l],
                                                    m1b[l], outbuf[l & 1], nullptr, NN, 0);
        xcur = outbuf[l & 1];
    }

    // ---- heads ----
    float* out_nodes = (float*)d_out;
    float* out_edges = (float*)d_out + NN;
    gemm_bf16<<<cdiv(NN, 64), 256, 0, stream>>>(xcur, nullptr, 128, 128, w_nc0, nc0_b,
                                                hmid, nullptr, NN, 1);
    node_logit<<<cdiv(NN * 8, 256), 256, 0, stream>>>(hmid, nc1_w, nc1_b, out_nodes);
    edge_cls<<<cdiv(NE2, 64), 256, 0, stream>>>(xcur, eidx, w_ec0, ec0_b, ec1_w, ec1_b,
                                                out_edges);

    // ---- group softmax over concatenated logits ----
    fill_f32<<<1, 256, 0, stream>>>(gmax, -INFINITY, GG);
    fill_f32<<<1, 256, 0, stream>>>(gsum, 0.f, GG);
    int T = NN + NE2;
    grp_max<<<cdiv(T, 256), 256, 0, stream>>>((float*)d_out, ibatch, eidx, gmax);
    grp_exp<<<cdiv(T, 256), 256, 0, stream>>>((float*)d_out, ibatch, eidx, gmax, gsum);
    grp_norm<<<cdiv(T, 256), 256, 0, stream>>>((float*)d_out, ibatch, eidx, gsum);
}
